// TransformerEncoderLayer_equi_an_21603685499585
// MI455X (gfx1250) — matrix operand order repaired and checked by reading, not compiled, not run on hardware
//
#include <hip/hip_runtime.h>

// ---------------- problem constants ----------------
#define CP   20000
#define CN   16
#define CD   128
#define CH   8
#define CAN  5
#define CF   512
#define CDA  640          // D * AN
#define PT   16           // points per workgroup -> 80 columns = 5 N-tiles
#define NCOL 80

// LDS padded strides (f16 elems): 16B-per-lane B reads land on distinct banks
#define XSTR 136          // 272B row = 17*16
#define YSTR 136
#define HSTR 520          // 1040B row = 65*16
#define OSTR 129          // fp32 LN buffer, odd stride -> conflict-free columns

typedef __attribute__((ext_vector_type(16))) _Float16 v16h;
typedef __attribute__((ext_vector_type(8)))  _Float16 v8h;
typedef __attribute__((ext_vector_type(8)))  float    v8f;

// ---- CDNA5 async global<->LDS copies (ASYNCcnt-tracked, per-lane LDS addr) ----
__device__ __forceinline__ void async_b128(void* lds, const void* gaddr) {
  uint32_t l = (uint32_t)(uintptr_t)lds;   // low 32 bits of generic LDS ptr = LDS offset
  asm volatile("global_load_async_to_lds_b128 %0, %1, off"
               :: "v"(l), "v"(gaddr) : "memory");
}
__device__ __forceinline__ void async_store_b128(void* gaddr, const void* lds) {
  uint32_t l = (uint32_t)(uintptr_t)lds;
  asm volatile("global_store_async_from_lds_b128 %0, %1, off"
               :: "v"(gaddr), "v"(l) : "memory");
}
__device__ __forceinline__ void async_wait0() {
  asm volatile("s_wait_asynccnt 0x0" ::: "memory");
}

__device__ __forceinline__ float wave_sum(float x) {
#pragma unroll
  for (int off = 16; off > 0; off >>= 1) x += __shfl_xor(x, off, 32);
  return x;
}

// A operand (16x32 f16), ISA layout: lane L -> row = L%16, half = L/16
__device__ __forceinline__ v16h load_a16(const _Float16* __restrict__ A, int lda,
                                         int row0, int k0) {
  const int lane = threadIdx.x & 31;
  const int r = lane & 15, hf = lane >> 4;
  const _Float16* p = A + (size_t)(row0 + r) * lda + k0 + hf * 8;
  v8h lo = *(const v8h*)(p);
  v8h hi = *(const v8h*)(p + 16);
  v16h out;
#pragma unroll
  for (int i = 0; i < 8; ++i) { out[i] = lo[i]; out[i + 8] = hi[i]; }
  return out;
}

// B operand (32x16 f16) from per-column-contiguous-K storage Bc[col][K]:
// lane L -> col = L%16, half = L/16, holds K = k0 + half*8 + [0..7] and k0 + 16 + half*8 + [0..7]
__device__ __forceinline__ v16h load_b16(const _Float16* __restrict__ Bc, int ldb,
                                         int col0, int k0) {
  const int lane = threadIdx.x & 31;
  const int c = lane & 15, hf = lane >> 4;
  const _Float16* p = Bc + (size_t)(col0 + c) * ldb + k0 + hf * 8;
  v8h lo = *(const v8h*)(p);
  v8h hi = *(const v8h*)(p + 16);
  v16h out;
#pragma unroll
  for (int i = 0; i < 8; ++i) { out[i] = lo[i]; out[i + 8] = hi[i]; }
  return out;
}

__device__ __forceinline__ v8f wmma16(v16h a, v16h b, v8f c) {
  return __builtin_amdgcn_wmma_f32_16x16x32_f16(false, a, false, b, (short)0, c,
                                                false, false);
}

// ---------------- K0: fp32 -> f16 weight conversion ----------------
__global__ void k_cvt(const float* __restrict__ src, _Float16* __restrict__ dst, int n) {
  int i = blockIdx.x * blockDim.x + threadIdx.x;
  if (i < n) dst[i] = (_Float16)src[i];
}

// ---------------- K1: Q = Wq*X, K = Wk*X (V == Q in the reference) -------------
// q/k stored f16 [P][d*5+a]; C-tiles scattered to LDS then async-streamed out.
__global__ __launch_bounds__(256) void k_proj(const float* __restrict__ tgt,
                                              const _Float16* __restrict__ Wq16,
                                              const _Float16* __restrict__ Wk16,
                                              _Float16* __restrict__ qh,
                                              _Float16* __restrict__ kh) {
  __shared__ __align__(16) char smem[40960 + 2 * NCOL * XSTR];
  float*    xraw = (float*)smem;                    // 40960 B stage
  _Float16* Xt   = (_Float16*)(smem + 40960);       // 21760 B B-operand tile
  _Float16* qt   = (_Float16*)smem;                 // alias xraw: 20480 B out tile
  _Float16* kt   = (_Float16*)(smem + 20480);       // 20480 B out tile
  const int p0 = blockIdx.x * PT;
  const int tid = threadIdx.x;

  for (int j = tid; j < PT * CDA / 4; j += 256)     // 2560 x 16B chunks
    async_b128(&xraw[j * 4], tgt + (size_t)p0 * CDA + j * 4);
  async_wait0();
  __syncthreads();

  for (int j = tid; j < NCOL * CD; j += 256) {
    int col = j >> 7, e = j & 127;
    int lp = col / CAN, a = col - lp * CAN;         // col = lp*5 + a
    Xt[col * XSTR + e] = (_Float16)xraw[lp * CDA + e * CAN + a];
  }
  __syncthreads();                                  // xraw dead -> qt/kt live

  const int wave = tid >> 5, lane = tid & 31;
  const int d0 = wave * 16;
  const int r15 = lane & 15, hf = lane >> 4;
  for (int nt = 0; nt < 5; ++nt) {
    v8f accq = {}; v8f acck = {};
#pragma unroll
    for (int kt2 = 0; kt2 < 4; ++kt2) {
      v16h b  = load_b16(Xt, XSTR, nt * 16, kt2 * 32);
      v16h aq = load_a16(Wq16, CD, d0, kt2 * 32);
      v16h ak = load_a16(Wk16, CD, d0, kt2 * 32);
      accq = wmma16(aq, b, accq);
      acck = wmma16(ak, b, acck);
    }
    int col = nt * 16 + r15;
    int lp = col / CAN, a = col - lp * CAN;
#pragma unroll
    for (int r = 0; r < 8; ++r) {
      int d = d0 + r + hf * 8;
      qt[lp * CDA + d * CAN + a] = (_Float16)accq[r];   // cheap LDS scatter
      kt[lp * CDA + d * CAN + a] = (_Float16)acck[r];
    }
  }
  __syncthreads();

  // stream out: 16 rows x 80 x 16B chunks per array
  for (int j = tid; j < PT * 80; j += 256) {
    int row = j / 80, c = j - row * 80;
    size_t gb = (size_t)(p0 + row) * CDA + c * 8;
    async_store_b128(qh + gb, &qt[row * CDA + c * 8]);
    async_store_b128(kh + gb, &kt[row * CDA + c * 8]);
  }
  async_wait0();
}

// ---------------- K2: irregular neighbor attention ----------------
__global__ __launch_bounds__(256) void k_attn(const _Float16* __restrict__ qh,
                                              const _Float16* __restrict__ kh,
                                              const int* __restrict__ idx,
                                              const float* __restrict__ dist,
                                              _Float16* __restrict__ t2h) {
  __shared__ float    qv[CDA];
  __shared__ _Float16 kn[CN][CDA];           // 20 KB
  __shared__ _Float16 vn[CN][CDA];           // 20 KB
  __shared__ _Float16 t2s[CDA];              // staged output row [a][d]
  __shared__ int   nb[CN];
  __shared__ float sc[CN][CH];
  __shared__ float att[CN][CH];
  const int p = blockIdx.x;
  const int tid = threadIdx.x;

  if (tid < CN) nb[tid] = idx[p * CN + tid];
  for (int j = tid; j < CDA; j += 256) qv[j] = (float)qh[(size_t)p * CDA + j];
  __syncthreads();

  for (int j = tid; j < CN * 80; j += 256) { // 16 rows x 80 x 16B chunks per array
    int n = j / 80, c = j - n * 80;
    size_t gb = (size_t)nb[n] * CDA + c * 8;
    async_b128(&kn[n][c * 8], kh + gb);
    async_b128(&vn[n][c * 8], qh + gb);      // value path uses Q projection (faithful)
  }
  async_wait0();
  __syncthreads();

  if (tid < CN * CH) {                       // 128 threads: one (n,h) each
    int n = tid >> 3, h = tid & 7;
    int base = h * 80;                       // (h*16+c)*5+i contiguous [h*80, h*80+80)
    float s = 0.f;
#pragma unroll 8
    for (int j = 0; j < 80; ++j) s += qv[base + j] * (float)kn[n][base + j];
    sc[n][h] = (s + dist[((size_t)p * CN + n) * CH + h]) * 0.25f;  // 1/sqrt(DH)
  }
  __syncthreads();

  if (tid < CH) {
    int h = tid;
    float m = -1e30f;
    for (int n = 0; n < CN; ++n) m = fmaxf(m, sc[n][h]);
    float e[CN], ssum = 0.f;
    for (int n = 0; n < CN; ++n) { e[n] = __expf(sc[n][h] - m); ssum += e[n]; }
    float inv = 1.f / ssum;
    for (int n = 0; n < CN; ++n) att[n][h] = e[n] * inv;
  }
  __syncthreads();

  for (int dd = tid; dd < CDA; dd += 256) {  // dd = e*5 + a, head h = dd/80
    int h = dd / 80;
    float s = 0.f;
#pragma unroll
    for (int n = 0; n < CN; ++n) s += att[n][h] * (float)vn[n][dd];
    int e = dd / CAN, a = dd - e * CAN;
    t2s[a * CD + e] = (_Float16)s;           // LDS scatter, [AN][D] row
  }
  __syncthreads();

  if (tid < 80) async_store_b128(t2h + (size_t)p * CDA + tid * 8, &t2s[tid * 8]);
  async_wait0();
}

// ---------------- K3: O = Wo*T2 + x, LayerNorm1, emit y (f16 + f32) -----------
__global__ __launch_bounds__(256) void k_wo_ln(const _Float16* __restrict__ t2h,
                                               const float* __restrict__ tgt,
                                               const _Float16* __restrict__ Wo16,
                                               const float* __restrict__ g1,
                                               const float* __restrict__ be1,
                                               _Float16* __restrict__ y16,
                                               float* __restrict__ y32) {
  __shared__ _Float16 Tt[NCOL * XSTR];
  __shared__ float    O[NCOL * OSTR];
  const int p0 = blockIdx.x * PT;
  const int tid = threadIdx.x;

  // t2h already col-major-K: async scatter into padded LDS. 80 cols x 16 chunks.
  for (int j = tid; j < NCOL * 16; j += 256) {
    int col = j >> 4, w = j & 15;
    async_b128(&Tt[col * XSTR + w * 8],
               t2h + ((size_t)p0 * CAN + col) * CD + w * 8);
  }
  async_wait0();
  __syncthreads();

  const int wave = tid >> 5, lane = tid & 31;
  const int d0 = wave * 16;
  const int r15 = lane & 15, hf = lane >> 4;
  for (int nt = 0; nt < 5; ++nt) {
    v8f acc = {};
#pragma unroll
    for (int kt = 0; kt < 4; ++kt) {
      v16h a = load_a16(Wo16, CD, d0, kt * 32);
      v16h b = load_b16(Tt, XSTR, nt * 16, kt * 32);
      acc = wmma16(a, b, acc);
    }
    int col = nt * 16 + r15;
    int lp = col / CAN, a = col - lp * CAN;
    size_t xb = (size_t)(p0 + lp) * CDA + a;
#pragma unroll
    for (int r = 0; r < 8; ++r) {
      int d = d0 + r + hf * 8;
      O[col * OSTR + d] = acc[r] + tgt[xb + (size_t)d * CAN];   // residual
    }
  }
  __syncthreads();

  // wave-parallel LayerNorm: one wave per column, 4 elems/lane, shfl reductions
  for (int col = wave; col < NCOL; col += 8) {
    float x[4];
#pragma unroll
    for (int k = 0; k < 4; ++k) x[k] = O[col * OSTR + lane + 32 * k];
    float m = wave_sum(x[0] + x[1] + x[2] + x[3]) * (1.f / CD);
    float vs = 0.f;
#pragma unroll
    for (int k = 0; k < 4; ++k) { x[k] -= m; vs += x[k] * x[k]; }
    float rstd = rsqrtf(wave_sum(vs) * (1.f / CD) + 1e-5f);
    int lp = col / CAN, a = col - lp * CAN;
    size_t ob = ((size_t)(p0 + lp) * CAN + a) * CD;   // [P, AN, D]
#pragma unroll
    for (int k = 0; k < 4; ++k) {
      int d = lane + 32 * k;
      float yv = x[k] * rstd * g1[d] + be1[d];
      y32[ob + d] = yv;                    // coalesced 128B runs per lane-group
      y16[ob + d] = (_Float16)yv;
    }
  }
}

// ---------------- K4: fused FFN (both GEMMs) + LayerNorm2 + transpose out -----
__global__ __launch_bounds__(256) void k_ffn(const _Float16* __restrict__ y16,
                                             const float* __restrict__ y32,
                                             const _Float16* __restrict__ W116,
                                             const _Float16* __restrict__ W216,
                                             const float* __restrict__ b1,
                                             const float* __restrict__ b2,
                                             const float* __restrict__ g2,
                                             const float* __restrict__ be2,
                                             float* __restrict__ out) {
  __shared__ __align__(16) char smem[NCOL * YSTR * 2 + NCOL * HSTR * 2 + NCOL * OSTR * 4];
  _Float16* yls  = (_Float16*)smem;                                  // 21760 B
  _Float16* hid  = (_Float16*)(smem + NCOL * YSTR * 2);              // 83200 B
  float*    O    = (float*)(smem + NCOL * YSTR * 2 + NCOL * HSTR * 2); // 41280 B
  float*    Ofin = (float*)(smem + NCOL * YSTR * 2);                 // aliases hid
  const int p0 = blockIdx.x * PT;
  const int tid = threadIdx.x;

  for (int j = tid; j < NCOL * 16; j += 256) {     // stage y tile (padded scatter)
    int col = j >> 4, w = j & 15;
    async_b128(&yls[col * YSTR + w * 8],
               y16 + ((size_t)p0 * CAN + col) * CD + w * 8);
  }
  async_wait0();
  __syncthreads();

  const int wave = tid >> 5, lane = tid & 31;
  const int r15 = lane & 15, hf = lane >> 4;

  // GEMM1: hidden = relu(W1 (512x128) * Y (128x80) + b1)
  for (int nt = 0; nt < 5; ++nt) {
    v16h bk0 = load_b16(yls, YSTR, nt * 16, 0);
    v16h bk1 = load_b16(yls, YSTR, nt * 16, 32);
    v16h bk2 = load_b16(yls, YSTR, nt * 16, 64);
    v16h bk3 = load_b16(yls, YSTR, nt * 16, 96);
#pragma unroll
    for (int mt = 0; mt < 4; ++mt) {
      int f0 = (wave * 4 + mt) * 16;
      v8f acc = {};
      acc = wmma16(load_a16(W116, CD, f0,  0), bk0, acc);
      acc = wmma16(load_a16(W116, CD, f0, 32), bk1, acc);
      acc = wmma16(load_a16(W116, CD, f0, 64), bk2, acc);
      acc = wmma16(load_a16(W116, CD, f0, 96), bk3, acc);
      int col = nt * 16 + r15;
#pragma unroll
      for (int r = 0; r < 8; ++r) {
        int f = f0 + r + hf * 8;
        hid[col * HSTR + f] = (_Float16)fmaxf(acc[r] + b1[f], 0.f);
      }
    }
  }
  __syncthreads();

  // GEMM2: O = W2 (128x512) * hidden (512x80) + b2 + y  (residual)
  const int d0 = wave * 16;
  for (int nt = 0; nt < 5; ++nt) {
    v8f acc = {};
#pragma unroll 4
    for (int kt = 0; kt < 16; ++kt) {
      v16h a = load_a16(W216, CF, d0, kt * 32);
      v16h b = load_b16(hid, HSTR, nt * 16, kt * 32);
      acc = wmma16(a, b, acc);
    }
    int col = nt * 16 + r15;
    int lp = col / CAN, aa = col - lp * CAN;
    size_t yb = ((size_t)(p0 + lp) * CAN + aa) * CD;
#pragma unroll
    for (int r = 0; r < 8; ++r) {
      int d = d0 + r + hf * 8;
      O[col * OSTR + d] = acc[r] + b2[d] + y32[yb + d];
    }
  }
  __syncthreads();                                  // hid dead -> Ofin live

  // wave-parallel LayerNorm2, scatter result into [p][d*5+a] LDS tile
  for (int col = wave; col < NCOL; col += 8) {
    float x[4];
#pragma unroll
    for (int k = 0; k < 4; ++k) x[k] = O[col * OSTR + lane + 32 * k];
    float m = wave_sum(x[0] + x[1] + x[2] + x[3]) * (1.f / CD);
    float vs = 0.f;
#pragma unroll
    for (int k = 0; k < 4; ++k) { x[k] -= m; vs += x[k] * x[k]; }
    float rstd = rsqrtf(wave_sum(vs) * (1.f / CD) + 1e-5f);
    int lp = col / CAN, a = col - lp * CAN;
#pragma unroll
    for (int k = 0; k < 4; ++k) {
      int d = lane + 32 * k;
      Ofin[lp * CDA + d * CAN + a] = x[k] * rstd * g2[d] + be2[d];
    }
  }
  __syncthreads();

  // contiguous async stream-out: 16 rows x 2560B = 2560 x 16B chunks
  for (int j = tid; j < PT * CDA / 4; j += 256)
    async_store_b128(out + (size_t)p0 * CDA + j * 4, &Ofin[j * 4]);
  async_wait0();
}

extern "C" void kernel_launch(void* const* d_in, const int* in_sizes, int n_in,
                              void* d_out, int out_size, void* d_ws, size_t ws_size,
                              hipStream_t stream) {
  const float* tgt  = (const float*)d_in[0];
  const int*   idx  = (const int*)d_in[1];
  const float* dist = (const float*)d_in[3];
  const float* Wq   = (const float*)d_in[4];
  const float* Wk   = (const float*)d_in[5];
  const float* Wo   = (const float*)d_in[6];
  const float* W1   = (const float*)d_in[7];
  const float* b1   = (const float*)d_in[8];
  const float* W2   = (const float*)d_in[9];
  const float* b2   = (const float*)d_in[10];
  const float* g1   = (const float*)d_in[11];
  const float* be1  = (const float*)d_in[12];
  const float* g2   = (const float*)d_in[13];
  const float* be2  = (const float*)d_in[14];

  char* ws = (char*)d_ws;
  _Float16* Wq16 = (_Float16*)ws;
  _Float16* Wk16 = Wq16 + CD * CD;
  _Float16* Wo16 = Wk16 + CD * CD;
  _Float16* W116 = Wo16 + CD * CD;
  _Float16* W216 = W116 + CF * CD;

  const size_t big = (size_t)CP * CDA;       // 12.8M elements per buffer
  _Float16* qh  = (_Float16*)(ws + (1 << 20));   // 25.6 MB
  _Float16* kh  = qh + big;                      // 25.6 MB
  _Float16* t2h = kh + big;                      // 25.6 MB
  _Float16* y16 = qh;                            // alias: q dead after attention
  float*    y32 = (float*)(t2h + big);           // 51.2 MB, fresh region

  k_cvt<<<(CD * CD + 255) / 256, 256, 0, stream>>>(Wq, Wq16, CD * CD);
  k_cvt<<<(CD * CD + 255) / 256, 256, 0, stream>>>(Wk, Wk16, CD * CD);
  k_cvt<<<(CD * CD + 255) / 256, 256, 0, stream>>>(Wo, Wo16, CD * CD);
  k_cvt<<<(CF * CD + 255) / 256, 256, 0, stream>>>(W1, W116, CF * CD);
  k_cvt<<<(CF * CD + 255) / 256, 256, 0, stream>>>(W2, W216, CF * CD);

  k_proj<<<CP / PT, 256, 0, stream>>>(tgt, Wq16, Wk16, qh, kh);
  k_attn<<<CP, 256, 0, stream>>>(qh, kh, idx, dist, t2h);
  k_wo_ln<<<CP / PT, 256, 0, stream>>>(t2h, tgt, Wo16, g1, be1, y16, y32);
  k_ffn<<<CP / PT, 256, 0, stream>>>(y16, y32, W116, W216, b1, b2, g2, be2,
                                     (float*)d_out);
}
